// MEL_Spectrogram_58987080843817
// MI455X (gfx1250) — matrix operand order repaired and checked by reading, not compiled, not run on hardware
//
#include <hip/hip_runtime.h>
#include <hip/hip_bf16.h>

typedef __attribute__((ext_vector_type(16))) _Float16 v16h;
typedef __attribute__((ext_vector_type(8)))  float    v8f;
typedef __attribute__((ext_vector_type(4)))  _Float16 h4;

#define WS_     1024
#define HOP_    512
#define NMEL    128
#define CUT     513            // WS/2 + 1
#define NSIG    524288
#define TOUT    1025           // output time frames (mag[...,1:-1])
#define MTILE   32             // frames per workgroup
#define NTILT   33             // ceil(1025/32)
#define SA_LEN  16896          // (MTILE-1)*HOP_ + WS_ unique samples per tile

// ---------------------------------------------------------------------------
// Prep 1: mel_filter rows are one-hot -> extract bin index + weight per row.
// ---------------------------------------------------------------------------
__global__ void mel_extract_bins(const float* __restrict__ mf,
                                 int* __restrict__ bins,
                                 float* __restrict__ wts) {
    int m = threadIdx.x;
    if (m >= NMEL) return;
    int bin = 0; float w = 0.0f;
    for (int f = 0; f < CUT; ++f) {
        float v = mf[m * CUT + f];
        if (v != 0.0f) { bin = f; w = v; }
    }
    bins[m] = bin;
    wts[m]  = w;
}

// ---------------------------------------------------------------------------
// Prep 2: pack the 256 needed basis rows (128 re + 128 im) into f16 in the
// exact WMMA 16-bit B-fragment layout (ISA 7.12.2):
//   lane L: N = L%16 ; half = L/16 ; VGPR r holds K = 16*(r/4) + half*8 + 2*(r%4), +1
// Packed index: [tile j (0..15)][kstep (0..31)][lane (0..31)][16 f16]
// ---------------------------------------------------------------------------
__global__ void mel_pack_basis(const float* __restrict__ fb,
                               const int* __restrict__ bins,
                               _Float16* __restrict__ Bpack) {
    int gid  = blockIdx.x * blockDim.x + threadIdx.x;   // 0 .. 16383
    int lane = gid & 31;
    int ks   = (gid >> 5) & 31;
    int j    = gid >> 10;                               // N-tile 0..15
    int n    = j * 16 + (lane & 15);                    // 0..255
    int row  = (n < NMEL) ? bins[n] : (CUT + bins[n - NMEL]);
    const float* src = fb + (size_t)row * WS_;
    _Float16*    dst = Bpack + (size_t)gid * 16;
#pragma unroll
    for (int e = 0; e < 16; ++e) {
        int r = e >> 1, h = e & 1;
        int k = ks * 32 + ((e >> 3) << 4) + ((lane >> 4) << 3) + (e & 7);  // 16*(e/8) + half*8 + e%8
        dst[e] = (_Float16)src[k];
    }
}

// ---- fragment load helpers -------------------------------------------------
__device__ __forceinline__ v16h load_afrag(const _Float16* ap) {
    // 16-bit A layout: this lane-half needs k = {0..7} and {16..23} relative
    // to ap -> exactly two aligned 16-byte LDS loads.
    union { uint4 u[2]; v16h h; } A;
    A.u[0] = *(const uint4*)(const void*)(ap);
    A.u[1] = *(const uint4*)(const void*)(ap + 16);
    return A.h;
}

__device__ __forceinline__ v16h load_bfrag(const _Float16* bp) {
    union { uint4 u[2]; v16h h; } B;
    B.u[0] = *(const uint4*)(const void*)(bp);
    B.u[1] = *(const uint4*)(const void*)(bp + 8);
    return B.h;
}

// ---------------------------------------------------------------------------
// Main: per-(batch, 32-frame tile) GEMM via v_wmma_f32_16x16x32_f16.
// 8 waves: mtile = wave>>2 selects frames 0-15 / 16-31,
//          q = wave&3 selects N-tiles {q, q+4, q+8, q+12} so each wave holds
//          matching (re, im) accumulator pairs for magnitude.
// K loop is software-pipelined with prefetch distance 2; sched_barrier(0)
// fences pin the load groups so the scheduler cannot re-serialize them onto
// their consuming WMMAs (keeps s_wait_loadcnt > 0 in steady state).
// ---------------------------------------------------------------------------
__global__ void __launch_bounds__(256)
mel_spectrogram_wmma(const float* __restrict__ sig,
                     const _Float16* __restrict__ Bpack,
                     const float* __restrict__ wts,
                     float* __restrict__ out) {
    __shared__ _Float16 sA[SA_LEN];          // 33,792 B: overlap-shared frames

    const int b  = blockIdx.y;
    const int T0 = blockIdx.x * MTILE;
    const float* s = sig + (size_t)b * NSIG;

    // ---- stage A: samples T0*512-512 .. +16895, f32 -> f16, OOB = 0 -------
    const int base = T0 * HOP_ - HOP_;       // first needed sample (may be <0)
    for (int i4 = threadIdx.x; i4 < SA_LEN / 4; i4 += 256) {
        int g = base + i4 * 4;               // multiple of 4 -> whole vec in/out
        float4 v = make_float4(0.f, 0.f, 0.f, 0.f);
        if (g >= 0 && g < NSIG) v = *(const float4*)(s + g);
        h4 hv = { (_Float16)v.x, (_Float16)v.y, (_Float16)v.z, (_Float16)v.w };
        *(h4*)(&sA[i4 * 4]) = hv;
    }
    __syncthreads();

    const int lane  = threadIdx.x & 31;
    const int wave  = threadIdx.x >> 5;
    const int q     = wave & 3;              // N group
    const int mtile = wave >> 2;             // 0/1: frames 0-15 / 16-31
    const int hi16  = lane >> 4;             // lane half
    const int col   = lane & 15;

    const int frame = mtile * 16 + col;      // A layout: lane L -> M = L%16
    const _Float16* sAf = &sA[frame * HOP_] + hi16 * 8;
    const _Float16* Bp  = Bpack + ((size_t)(q * 32) * 32 + lane) * 16;
    const size_t BT = (size_t)4 * 32 * 32 * 16;   // +4 N-tiles stride (f16)
    const size_t BK = (size_t)32 * 16;            // k-step stride (f16)

    v8f acc[4];
#pragma unroll
    for (int p = 0; p < 4; ++p) acc[p] = (v8f){};

    // ---- software-pipelined K loop, prefetch distance 2 --------------------
    v16h Af[2];
    v16h Bf[2][4];
    Af[0] = load_afrag(sAf);
#pragma unroll
    for (int p = 0; p < 4; ++p) Bf[0][p] = load_bfrag(Bp + p * BT);
    Af[1] = load_afrag(sAf + 32);
#pragma unroll
    for (int p = 0; p < 4; ++p) Bf[1][p] = load_bfrag(Bp + BK + p * BT);
    __builtin_amdgcn_sched_barrier(0);

#pragma unroll
    for (int ks = 0; ks < 32; ++ks) {
        const int cur = ks & 1;
        // compute step ks: its loads were issued 2 iterations ago, so the
        // inserted wait is loadcnt <= (next step's 8 loads), not 0.
#pragma unroll
        for (int p = 0; p < 4; ++p)
            acc[p] = __builtin_amdgcn_wmma_f32_16x16x32_f16(
                false, Af[cur], false, Bf[cur][p], (short)0, acc[p], false, false);
        __builtin_amdgcn_sched_barrier(0);
        if (ks + 2 < 32) {
            // refill the buffer just consumed with step ks+2
            Af[cur] = load_afrag(sAf + (ks + 2) * 32);
#pragma unroll
            for (int p = 0; p < 4; ++p)
                Bf[cur][p] = load_bfrag(Bp + (ks + 2) * BK + p * BT);
        }
        __builtin_amdgcn_sched_barrier(0);
    }

    // ---- magnitude + mel weight + log, store (C/D layout: lane=N, VGPR=M) --
    union { v8f v; float f[8]; } R0, R1, I0, I1;
    R0.v = acc[0]; R1.v = acc[1]; I0.v = acc[2]; I1.v = acc[3];

    const float w0 = wts[q * 16 + col];
    const float w1 = wts[(q + 4) * 16 + col];
    const int   m0 = q * 16 + col;
    const int   m1 = (q + 4) * 16 + col;
    float* outb = out + (size_t)b * NMEL * TOUT;

#pragma unroll
    for (int r = 0; r < 8; ++r) {
        int t = T0 + mtile * 16 + hi16 * 8 + r;
        if (t < TOUT) {
            float g0 = sqrtf(R0.f[r] * R0.f[r] + I0.f[r] * I0.f[r]);
            float g1 = sqrtf(R1.f[r] * R1.f[r] + I1.f[r] * I1.f[r]);
            outb[(size_t)m0 * TOUT + t] = logf(w0 * g0 + 1e-10f);
            outb[(size_t)m1 * TOUT + t] = logf(w1 * g1 + 1e-10f);
        }
    }
}

// ---------------------------------------------------------------------------
extern "C" void kernel_launch(void* const* d_in, const int* in_sizes, int n_in,
                              void* d_out, int out_size, void* d_ws, size_t ws_size,
                              hipStream_t stream) {
    const float* signal = (const float*)d_in[0];   // (32, 524288) f32
    const float* fb     = (const float*)d_in[1];   // (1026, 1024) f32
    const float* mf     = (const float*)d_in[2];   // (128, 513)   f32
    float* out = (float*)d_out;                    // (32, 128, 1025) f32

    // workspace layout: bins[128] | wts[128] | Bpack (16*32*32*16 f16 = 512KB)
    char* wsb = (char*)d_ws;
    int*      bins  = (int*)wsb;
    float*    wts   = (float*)(wsb + 512);
    _Float16* Bpack = (_Float16*)(wsb + 1024);

    mel_extract_bins<<<1, 128, 0, stream>>>(mf, bins, wts);
    mel_pack_basis<<<64, 256, 0, stream>>>(fb, bins, Bpack);
    mel_spectrogram_wmma<<<dim3(NTILT, 32), 256, 0, stream>>>(signal, Bpack, wts, out);
}
